// Restorer_66340064854390
// MI455X (gfx1250) — matrix operand order repaired and checked by reading, not compiled, not run on hardware
//
#include <hip/hip_runtime.h>
#include <hip/hip_bf16.h>

static constexpr int VD   = 512;   // vocab V
static constexpr int BB   = 64;    // batch B
static constexpr int HH   = 128;   // seq H
static constexpr int BH   = BB * HH;
static constexpr float EPS_FIN = 1.1920928955078125e-07f; // float32 eps

typedef __attribute__((ext_vector_type(16))) _Float16 v16h;
typedef __attribute__((ext_vector_type(8)))  _Float16 v8h;
typedef __attribute__((ext_vector_type(8)))  float    v8f;
typedef __attribute__((ext_vector_type(4)))  unsigned int u32x4;
typedef __attribute__((ext_vector_type(4)))  int          i32x4;
typedef __attribute__((ext_vector_type(8)))  int          i32x8;

// ---------------------------------------------------------------------------
// Kernel 1: row softmax over V; outputs p (f32+f16), log(p+1e-6) (f16), nll row
// One 256-thread block per row (bh). Each thread owns elements tid and tid+256.
// ---------------------------------------------------------------------------
__global__ void softmax_kernel(const float* __restrict__ logits,
                               const int*   __restrict__ xs,
                               float*       __restrict__ p32,
                               _Float16*    __restrict__ p16,
                               _Float16*    __restrict__ logp16,
                               float*       __restrict__ nll_row) {
    const int row = blockIdx.x;
    const int tid = threadIdx.x;
    const float* x = logits + (size_t)row * VD;
    __shared__ float red[256];

    float x0 = x[tid];
    float x1 = x[tid + 256];
    float m  = fmaxf(x0, x1);
    red[tid] = m; __syncthreads();
    for (int s = 128; s > 0; s >>= 1) {
        if (tid < s) red[tid] = fmaxf(red[tid], red[tid + s]);
        __syncthreads();
    }
    m = red[0]; __syncthreads();

    float e0 = __expf(x0 - m), e1 = __expf(x1 - m);
    red[tid] = e0 + e1; __syncthreads();
    for (int s = 128; s > 0; s >>= 1) {
        if (tid < s) red[tid] += red[tid + s];
        __syncthreads();
    }
    float sum = red[0];
    float inv = 1.0f / sum;
    float p0 = e0 * inv, p1 = e1 * inv;

    size_t base = (size_t)row * VD;
    p32[base + tid]        = p0;
    p32[base + tid + 256]  = p1;
    p16[base + tid]        = (_Float16)p0;
    p16[base + tid + 256]  = (_Float16)p1;
    logp16[base + tid]       = (_Float16)__logf(p0 + 1e-6f);
    logp16[base + tid + 256] = (_Float16)__logf(p1 + 1e-6f);

    if (tid == 0) {
        // log_softmax(x+eps)[xs] = x[xs] - lse(x)  (eps shift cancels)
        float lse = m + __logf(sum);
        nll_row[row] = lse - x[xs[row]];
    }
}

// ---------------------------------------------------------------------------
// Pack matrices[ts[b]-1] (f32, row-major [c][v]) into f16 WMMA B-fragments.
// B-fragment layout (v_wmma_*_16x16x32_f16, 32x16 K-by-N, contiguous k order used by this file):
//   lane l: n = l & 15;  element e (0..15): K = (l>>4)*16 + e
// Packed: Bp[((b*16 + kTile)*32 + nTile)*32 + lane][16 halves], lane-contiguous.
// ---------------------------------------------------------------------------
__global__ void pack_M_kernel(const float* __restrict__ matrices,
                              const int*   __restrict__ ts,
                              _Float16*    __restrict__ Bp) {
    int wave = threadIdx.x >> 5;
    int lane = threadIdx.x & 31;
    int fragGlobal = blockIdx.x * 8 + wave;   // 0 .. BB*512-1
    int b    = fragGlobal >> 9;               // / 512
    int frag = fragGlobal & 511;              // kTile*32 + nTile
    int kTile = frag >> 5;
    int nTile = frag & 31;
    int t = ts[b];
    const float* M = matrices + (size_t)(t - 1) * VD * VD;
    int c  = nTile * 16 + (lane & 15);        // N index (output col)
    int v0 = kTile * 32 + (lane >> 4) * 16;   // K start (16 consecutive v)
    const float* src = M + (size_t)c * VD + v0;
    _Float16* dst = Bp + ((size_t)fragGlobal * 32 + lane) * 16;
#pragma unroll
    for (int e = 0; e < 16; ++e) dst[e] = (_Float16)src[e];
}

// Same packing for the shared A matrix (S-GEMM B operand).
__global__ void pack_A_kernel(const float* __restrict__ A,
                              _Float16*    __restrict__ Ap) {
    int wave = threadIdx.x >> 5;
    int lane = threadIdx.x & 31;
    int frag = blockIdx.x * 8 + wave;         // 0..511
    int kTile = frag >> 5;
    int nTile = frag & 31;
    int c  = nTile * 16 + (lane & 15);
    int v0 = kTile * 32 + (lane >> 4) * 16;
    const float* src = A + (size_t)c * VD + v0;
    _Float16* dst = Ap + ((size_t)frag * 32 + lane) * 16;
#pragma unroll
    for (int e = 0; e < 16; ++e) dst[e] = (_Float16)src[e];
}

// ---------------------------------------------------------------------------
// WMMA GEMM with TDM-staged A strip.
//   out[row, n] = sum_k Ah[row, k] * B[k, n]
// Block = (batch, 32-row strip). The 32x512 f16 A strip (32 KB) is DMA'd into
// LDS once per block by the Tensor Data Mover (wave 0 issues; TDM ignores
// EXEC), then all 8 waves build A fragments from LDS. Each wave register-
// blocks 2 mTiles x 4 nTiles -> 128 v_wmma per wave.
// A-fragment (16x32 f16, same contiguous k order as packed B): lane l holds row m=l&15;
//   elements 0..15: K = (l>>4)*16 + e         (32B contiguous LDS chunk)
// ---------------------------------------------------------------------------
__global__ void gemm_kernel(const _Float16* __restrict__ Ah,
                            const _Float16* __restrict__ Bp,
                            float*          __restrict__ out,
                            int perBatchB) {
    __shared__ _Float16 ldsA[32 * VD];            // 32 KB A strip

    int wave = threadIdx.x >> 5;
    int lane = threadIdx.x & 31;
    int batch = blockIdx.x >> 2;                  // 0..63
    int mPair = blockIdx.x & 3;                   // 0..3 -> rows mPair*32..+31
    int rowStart = batch * HH + mPair * 32;

    // Keep the compiler from treating TDM-written LDS as never-stored.
    if (perBatchB == 2) ldsA[threadIdx.x] = (_Float16)0.0f;   // never taken

    if (wave == 0) {
        // Tensor DMA descriptor (D#): 2D tensor 32 rows x 512 f16, row stride
        // 512 elements; whole tensor is the tile; dest = ldsA.
        unsigned int lds_off = (unsigned int)(uintptr_t)(&ldsA[0]); // addr[31:0] = LDS offset
        unsigned long long ga =
            (unsigned long long)(uintptr_t)(Ah + (size_t)rowStart * VD);
        u32x4 g0;
        g0[0] = 1u;                                            // count=1 (valid), user mode
        g0[1] = lds_off;                                       // lds_addr
        g0[2] = (unsigned int)ga;                              // global_addr[31:0]
        g0[3] = ((unsigned int)(ga >> 32) & 0x01FFFFFFu)       // global_addr[56:32]
                | (2u << 30);                                  // type=2 ("image")
        i32x8 g1;
        g1[0] = (int)(1u << 16);         // data_size=1 -> 2 bytes; wg_mask=0
        g1[1] = (int)(512u << 16);       // tensor_dim0[15:0]=512 (bits 63:48)
        g1[2] = (int)(32u << 16);        // tensor_dim1[15:0]=32  (bits 95:80)
        g1[3] = (int)(512u << 16);       // tile_dim0=512         (bits 127:112)
        g1[4] = 32;                      // tile_dim1=32          (bits 143:128)
        g1[5] = 512;                     // tensor_dim0_stride[31:0]=512
        g1[6] = 0;
        g1[7] = 0;
        i32x4 gz  = {0, 0, 0, 0};        // groups 2/3 unused (2D tensor)
        i32x8 gz8 = {0, 0, 0, 0, 0, 0, 0, 0};
        __builtin_amdgcn_tensor_load_to_lds(g0, g1, gz, gz, gz8, 0);
        __builtin_amdgcn_s_wait_tensorcnt(0);
    }
    asm volatile("" ::: "memory");       // LDS now holds TDM-written data
    __syncthreads();

    int half = lane >> 4;
    int mrow = lane & 15;
    size_t bOff = perBatchB ? ((size_t)batch * 512 * 32 * 16) : 0;

    v8f acc[2][4] = {};
#pragma unroll
    for (int kT = 0; kT < 16; ++kT) {
        v16h af[2];
#pragma unroll
        for (int mi = 0; mi < 2; ++mi) {
            const _Float16* ap = &ldsA[(size_t)(mi * 16 + mrow) * VD + kT * 32 + half * 16];
            v8h a0 = *(const v8h*)ap;
            v8h a1 = *(const v8h*)(ap + 8);
            af[mi] = __builtin_shufflevector(a0, a1,
                0,1,2,3,4,5,6,7,8,9,10,11,12,13,14,15);
        }
#pragma unroll
        for (int ni = 0; ni < 4; ++ni) {
            int nTile = wave * 4 + ni;
            const _Float16* bPtr = Bp + bOff +
                (((size_t)kT * 32 + nTile) * 32 + lane) * 16;
            v8h b0 = *(const v8h*)(bPtr);
            v8h b1 = *(const v8h*)(bPtr + 8);
            v16h bf = __builtin_shufflevector(b0, b1,
                0,1,2,3,4,5,6,7,8,9,10,11,12,13,14,15);
            acc[0][ni] = __builtin_amdgcn_wmma_f32_16x16x32_f16(
                false, af[0], false, bf, (short)0, acc[0][ni], false, false);
            acc[1][ni] = __builtin_amdgcn_wmma_f32_16x16x32_f16(
                false, af[1], false, bf, (short)0, acc[1][ni], false, false);
        }
    }

    // D layout: lane holds column n = l&15; VGPR r -> row m = (l>>4)*8 + r
#pragma unroll
    for (int mi = 0; mi < 2; ++mi) {
#pragma unroll
        for (int ni = 0; ni < 4; ++ni) {
            int nTile = wave * 4 + ni;
            float* ob = out + (size_t)(rowStart + mi * 16) * VD
                            + nTile * 16 + (lane & 15);
#pragma unroll
            for (int r = 0; r < 8; ++r)
                ob[(size_t)(half * 8 + r) * VD] = acc[mi][ni][r];
        }
    }
}

// ---------------------------------------------------------------------------
// KL per row: EtXt gather from Q, true/pred distributions, KL element sum.
// ---------------------------------------------------------------------------
__global__ void kl_kernel(const float* __restrict__ Q,
                          const float* __restrict__ matrices,
                          const float* __restrict__ Em1,
                          const int*   __restrict__ xt,
                          const int*   __restrict__ xs,
                          const int*   __restrict__ ts,
                          float*       __restrict__ kl_row) {
    int row = blockIdx.x;
    int tid = threadIdx.x;
    int b = row >> 7;                 // / HH
    int t = ts[b];
    int xtv = xt[row], xsv = xs[row];
    const float* Qt = Q        + (size_t)t       * VD * VD;
    const float* Mt = matrices + (size_t)(t - 1) * VD * VD;
    __shared__ float redA[256], redB[256];

    float et0 = Qt[(size_t)tid * VD + xtv];
    float et1 = Qt[(size_t)(tid + 256) * VD + xtv];
    float tu0 = et0 * Mt[(size_t)tid * VD + xsv];
    float tu1 = et1 * Mt[(size_t)(tid + 256) * VD + xsv];
    float pu0 = et0 * Em1[(size_t)row * VD + tid];
    float pu1 = et1 * Em1[(size_t)row * VD + tid + 256];

    redA[tid] = tu0 + tu1;
    redB[tid] = pu0 + pu1;
    __syncthreads();
    for (int s = 128; s > 0; s >>= 1) {
        if (tid < s) { redA[tid] += redA[tid + s]; redB[tid] += redB[tid + s]; }
        __syncthreads();
    }
    float invT = 1.0f / redA[0];
    float invP = 1.0f / fmaxf(redB[0], 1e-8f);
    __syncthreads();

    float kl = 0.0f;
    {
        float tp = fminf(fmaxf(tu0 * invT, EPS_FIN), 1.0f - EPS_FIN);
        float pp = fminf(fmaxf(pu0 * invP, EPS_FIN), 1.0f - EPS_FIN);
        kl += tp * (__logf(tp) - (__logf(pp) + 1e-6f));
        tp = fminf(fmaxf(tu1 * invT, EPS_FIN), 1.0f - EPS_FIN);
        pp = fminf(fmaxf(pu1 * invP, EPS_FIN), 1.0f - EPS_FIN);
        kl += tp * (__logf(tp) - (__logf(pp) + 1e-6f));
    }
    redA[tid] = kl; __syncthreads();
    for (int s = 128; s > 0; s >>= 1) {
        if (tid < s) redA[tid] += redA[tid + s];
        __syncthreads();
    }
    if (tid == 0) kl_row[row] = redA[0];
}

// ---------------------------------------------------------------------------
// Consistency terms: t1[b,j] = <S[b,j+1,:], p[b,j,:]>, t2[b,j] = <S[b,j,:], p[b,j+1,:]>
// ---------------------------------------------------------------------------
__global__ void con_kernel(const float* __restrict__ S,
                           const float* __restrict__ p32,
                           float*       __restrict__ t12) {
    int b = blockIdx.x / (HH - 1);
    int j = blockIdx.x % (HH - 1);
    int tid = threadIdx.x;
    const float* Sj  = S   + ((size_t)b * HH + j) * VD;
    const float* Sj1 = Sj  + VD;
    const float* pj  = p32 + ((size_t)b * HH + j) * VD;
    const float* pj1 = pj  + VD;

    float a1 = Sj1[tid] * pj[tid]  + Sj1[tid + 256] * pj[tid + 256];
    float a2 = Sj[tid]  * pj1[tid] + Sj[tid + 256]  * pj1[tid + 256];
    __shared__ float rA[256], rB[256];
    rA[tid] = a1; rB[tid] = a2; __syncthreads();
    for (int s = 128; s > 0; s >>= 1) {
        if (tid < s) { rA[tid] += rA[tid + s]; rB[tid] += rB[tid + s]; }
        __syncthreads();
    }
    if (tid == 0) {
        t12[(size_t)blockIdx.x * 2]     = rA[0];
        t12[(size_t)blockIdx.x * 2 + 1] = rB[0];
    }
}

// ---------------------------------------------------------------------------
// Final reduction: masked, length-normalized sums -> 3 scalars.
// ---------------------------------------------------------------------------
__global__ void finalize_kernel(const float* __restrict__ kl_row,
                                const float* __restrict__ nll_row,
                                const float* __restrict__ t12,
                                const int*   __restrict__ lengths,
                                float*       __restrict__ out3) {
    int tid = threadIdx.x;
    float kl = 0.0f, ce = 0.0f, con = 0.0f;
    for (int row = tid; row < BH; row += 256) {
        int b = row >> 7, h = row & 127;
        int len = lengths[b];
        if (h < len) {
            float invL = 1.0f / (float)len;
            kl += kl_row[row]  * invL;
            ce += nll_row[row] * invL;
        }
    }
    for (int idx = tid; idx < BB * (HH - 1); idx += 256) {
        int b = idx / (HH - 1), j = idx % (HH - 1);
        int len = lengths[b];
        if (j < len - 1) {
            float denom = (float)(len - 1) * (float)VD;
            con += (t12[(size_t)idx * 2] + t12[(size_t)idx * 2 + 1]) / denom;
        }
    }
    __shared__ float rA[256], rB[256], rC[256];
    rA[tid] = kl; rB[tid] = ce; rC[tid] = con; __syncthreads();
    for (int s = 128; s > 0; s >>= 1) {
        if (tid < s) { rA[tid] += rA[tid+s]; rB[tid] += rB[tid+s]; rC[tid] += rC[tid+s]; }
        __syncthreads();
    }
    if (tid == 0) {
        out3[0] = rA[0];
        out3[1] = rB[0];
        out3[2] = (-rC[0] / (float)BB) * 100.0f;
    }
}

// ---------------------------------------------------------------------------
extern "C" void kernel_launch(void* const* d_in, const int* in_sizes, int n_in,
                              void* d_out, int out_size, void* d_ws, size_t ws_size,
                              hipStream_t stream) {
    const float* logits   = (const float*)d_in[0];  // (B,H,V)
    const float* matrices = (const float*)d_in[1];  // (T+1,V,V)
    const float* Q        = (const float*)d_in[2];  // (T+1,V,V)
    const float* A        = (const float*)d_in[3];  // (V,V)
    const int*   xs       = (const int*)d_in[4];    // (B,H)
    const int*   xt       = (const int*)d_in[5];    // (B,H)
    const int*   lengths  = (const int*)d_in[6];    // (B,)
    const int*   ts       = (const int*)d_in[7];    // (B,)
    (void)in_sizes; (void)n_in; (void)out_size; (void)ws_size;

    char* ws = (char*)d_ws;
    size_t off = 0;
    auto alloc = [&](size_t bytes) -> char* {
        char* p = ws + off;
        off += (bytes + 255) & ~(size_t)255;
        return p;
    };
    _Float16* p16    = (_Float16*)alloc((size_t)BH * VD * 2);
    _Float16* logp16 = (_Float16*)alloc((size_t)BH * VD * 2);
    float*    p32    = (float*)   alloc((size_t)BH * VD * 4);
    float*    Em1    = (float*)   alloc((size_t)BH * VD * 4);
    float*    Smat   = (float*)   alloc((size_t)BH * VD * 4);
    _Float16* Bp     = (_Float16*)alloc((size_t)BB * 512 * 32 * 16 * 2); // per-batch packed M
    _Float16* Ap     = (_Float16*)alloc((size_t)512 * 32 * 16 * 2);      // packed A
    float*    kl_row = (float*)   alloc((size_t)BH * 4);
    float*    nllrow = (float*)   alloc((size_t)BH * 4);
    float*    t12    = (float*)   alloc((size_t)BB * (HH - 1) * 2 * 4);

    softmax_kernel<<<BH, 256, 0, stream>>>(logits, xs, p32, p16, logp16, nllrow);
    pack_M_kernel<<<BB * 64, 256, 0, stream>>>(matrices, ts, Bp);
    pack_A_kernel<<<64, 256, 0, stream>>>(A, Ap);
    gemm_kernel<<<BB * 4, 256, 0, stream>>>(p16,    Bp, Em1,  1);
    gemm_kernel<<<BB * 4, 256, 0, stream>>>(logp16, Ap, Smat, 0);
    kl_kernel<<<BH, 256, 0, stream>>>(Q, matrices, Em1, xt, xs, ts, kl_row);
    con_kernel<<<BB * (HH - 1), 256, 0, stream>>>(Smat, p32, t12);
    finalize_kernel<<<1, 256, 0, stream>>>(kl_row, nllrow, t12, lengths, (float*)d_out);
}
